// HeteroGNN_68289980006597
// MI455X (gfx1250) — matrix operand order repaired and checked by reading, not compiled, not run on hardware
//
#include <hip/hip_runtime.h>
#include <math.h>

// ---------------------------------------------------------------------------
// Problem constants (match reference)
// ---------------------------------------------------------------------------
#define N_ADDR 100000
#define N_TX   150000
#define E_IN   500000
#define E_OUT  500000
#define E_SP   300000
#define F_ADDR 53
#define F_TX   6
#define HC     128      // H*C = 4*32
#define CH     32       // channels per head / node feature width

typedef __attribute__((ext_vector_type(16))) _Float16 v16h;
typedef __attribute__((ext_vector_type(8)))  float    v8f;

// ---------------------------------------------------------------------------
// GEMM:  Hout[N,128] = X[N,32] @ W[32,128]   (f32 in, f16 WMMA, f32 out)
// block = 256 threads = 8 waves; wave w -> column tile [16w,16w+16)
// each block covers 64 rows (4 sub-tiles of 16 per wave, B held in regs).
// FULL=true: whole 64-row block in range -> straight-line loads/stores.
// ---------------------------------------------------------------------------
template <bool FULL>
__device__ __forceinline__ void gemm_tile(const float* __restrict__ X,
                                          const float* __restrict__ W,
                                          float* __restrict__ Hout, int N,
                                          long base_row, int lane, int wv) {
  const int col0  = wv * 16;
  const int mlow  = lane & 15;
  const int khalf = lane >> 4;            // 0 or 1

  // B tile 32x16 f16: lane = column (lane&15); lane half selects K runs
  // {kh*8..+7, 16+kh*8..+7}; element j holds K = khalf*8 + j + (j & 8)
  v16h b;
  {
    const int n = col0 + mlow;
#pragma unroll
    for (int j = 0; j < 16; ++j)
      b[j] = (_Float16)W[(khalf * 8 + j + (j & 8)) * HC + n];
  }

#pragma unroll
  for (int t = 0; t < 4; ++t) {
    long row = base_row + t * 16 + mlow;
    long rr  = FULL ? row : ((row < N) ? row : (long)(N - 1)); // keep EXEC full
    const float4* xp = (const float4*)(X + rr * CH);
    // A 16x32 f16: lane = row; lane half selects K runs {kh*8..+7, 16+kh*8..+7}
    float4 r0 = xp[khalf * 2 + 0];
    float4 r1 = xp[khalf * 2 + 1];
    float4 r2 = xp[4 + khalf * 2 + 0];
    float4 r3 = xp[4 + khalf * 2 + 1];
    v16h a;
    a[0]=(_Float16)r0.x; a[1]=(_Float16)r0.y; a[2]=(_Float16)r0.z; a[3]=(_Float16)r0.w;
    a[4]=(_Float16)r1.x; a[5]=(_Float16)r1.y; a[6]=(_Float16)r1.z; a[7]=(_Float16)r1.w;
    a[8]=(_Float16)r2.x; a[9]=(_Float16)r2.y; a[10]=(_Float16)r2.z; a[11]=(_Float16)r2.w;
    a[12]=(_Float16)r3.x; a[13]=(_Float16)r3.y; a[14]=(_Float16)r3.z; a[15]=(_Float16)r3.w;

    v8f c = {0.f,0.f,0.f,0.f,0.f,0.f,0.f,0.f};
    c = __builtin_amdgcn_wmma_f32_16x16x32_f16(false, a, false, b,
                                               (short)0, c, false, false);
    // D layout: lane<16 -> N=lane, M=r ; lane>=16 -> N=lane-16, M=8+r
    const long m0 = base_row + t * 16 + 8 * khalf;
#pragma unroll
    for (int r = 0; r < 8; ++r) {
      if (FULL || (m0 + r) < N) Hout[(m0 + r) * HC + col0 + mlow] = c[r];
    }
  }
}

__global__ __launch_bounds__(256) void gemm_32x128(const float* __restrict__ X,
                                                   const float* __restrict__ W,
                                                   float* __restrict__ Hout,
                                                   int N) {
  const int lane = threadIdx.x & 31;
  const int wv   = threadIdx.x >> 5;       // 0..7 : column tile
  const long base_row = (long)blockIdx.x * 64;
  if (base_row + 64 <= (long)N)
    gemm_tile<true>(X, W, Hout, N, base_row, lane, wv);    // hot path, no guards
  else
    gemm_tile<false>(X, W, Hout, N, base_row, lane, wv);   // tail block only
}

// ---------------------------------------------------------------------------
// a[n,h] = sum_c Hfeat[n, h*32+c] * att[h*32+c]
// ---------------------------------------------------------------------------
__global__ __launch_bounds__(256) void attn_dot(const float* __restrict__ Hfeat,
                                                const float* __restrict__ att,
                                                float* __restrict__ out, int Nn) {
  int i = blockIdx.x * 256 + threadIdx.x;
  if (i >= Nn * 4) return;
  int n = i >> 2, h = i & 3;
  const float4* hp = (const float4*)(Hfeat + (long)n * HC + h * CH);
  const float4* ap = (const float4*)(att + h * CH);
  float s = 0.f;
#pragma unroll
  for (int c = 0; c < 8; ++c) {
    float4 hv = hp[c], av = ap[c];
    s += hv.x * av.x + hv.y * av.y + hv.z * av.z + hv.w * av.w;
  }
  out[i] = s;
}

// q[f,h] = sum_c w_edge[f, h*32+c] * att_edge[h*32+c]   (Fe*4 <= 12 values)
__global__ void edge_q(const float* __restrict__ w_edge,
                       const float* __restrict__ att_edge,
                       float* __restrict__ q, int Fe) {
  int i = threadIdx.x;
  if (i >= Fe * 4) return;
  int f = i >> 2, h = i & 3;
  float s = 0.f;
#pragma unroll
  for (int c = 0; c < CH; ++c) s += w_edge[f * HC + h * CH + c] * att_edge[h * CH + c];
  q[i] = s;
}

// per (edge,head): alpha -> leaky_relu -> exp; accumulate denom at dst
__global__ __launch_bounds__(256) void edge_alpha(const int* __restrict__ src,
                                                  const int* __restrict__ dst,
                                                  const float* __restrict__ a_src,
                                                  const float* __restrict__ a_dst,
                                                  const float* __restrict__ ea,
                                                  const float* __restrict__ q,
                                                  int Fe,
                                                  float* __restrict__ expal,
                                                  float* __restrict__ denom,
                                                  int E) {
  int i = blockIdx.x * 256 + threadIdx.x;
  if (i >= E * 4) return;
  int e = i >> 2, h = i & 3;
  float al = a_src[src[e] * 4 + h] + a_dst[dst[e] * 4 + h];
  if (ea != nullptr) {
    for (int f = 0; f < Fe; ++f) al += ea[(long)e * Fe + f] * q[f * 4 + h];
  }
  al = (al > 0.f) ? al : 0.2f * al;          // leaky_relu(0.2)
  float ex = __expf(al);                     // |al| tiny: safe without max-sub
  expal[i] = ex;
  unsafeAtomicAdd(&denom[dst[e] * 4 + h], ex);
}

// one thread per (edge, 4-channel quad): float4 gather + 4 f32 L2 atomics.
// total = E*32 ; quad q covers channels 4q..4q+3, head h = q>>3
__global__ __launch_bounds__(256) void scatter_msg(const int* __restrict__ src,
                                                   const int* __restrict__ dst,
                                                   const float* __restrict__ hs,
                                                   const float* __restrict__ expal,
                                                   float* __restrict__ acc,
                                                   long total) {
  long i = (long)blockIdx.x * 256 + threadIdx.x;
  if (i >= total) return;
  int e = (int)(i >> 5);
  int qd = (int)(i & 31);
  int h = qd >> 3;
  float w = expal[(long)e * 4 + h];
  const float4 v = *(const float4*)(hs + (long)src[e] * HC + qd * 4);
  float* ap = acc + (long)dst[e] * HC + qd * 4;
  unsafeAtomicAdd(ap + 0, v.x * w);
  unsafeAtomicAdd(ap + 1, v.y * w);
  unsafeAtomicAdd(ap + 2, v.z * w);
  unsafeAtomicAdd(ap + 3, v.w * w);
}

// out32[n,c] (+)= bias[c] + 0.25 * sum_h acc[n,h,c]/(denom[n,h]+1e-16)
__global__ __launch_bounds__(256) void finalize_gat(const float* __restrict__ acc,
                                                    const float* __restrict__ denom,
                                                    const float* __restrict__ bias,
                                                    float* __restrict__ out32,
                                                    int Nn, int accumulate) {
  int i = blockIdx.x * 256 + threadIdx.x;
  if (i >= Nn * CH) return;
  int n = i >> 5, c = i & 31;
  float s = 0.f;
#pragma unroll
  for (int h = 0; h < 4; ++h)
    s += acc[(long)n * HC + h * CH + c] / (denom[n * 4 + h] + 1e-16f);
  s = 0.25f * s + bias[c];
  out32[i] = (accumulate ? out32[i] : 0.f) + s;
}

// x = relu(out)+x ; block-reduce sum/sumsq into red[0..1]
__global__ __launch_bounds__(256) void resid_reduce(const float* __restrict__ o,
                                                    float* __restrict__ x,
                                                    float* __restrict__ red, long M) {
  __shared__ float s1[256], s2[256];
  long i = (long)blockIdx.x * 256 + threadIdx.x;
  float v = 0.f, v2 = 0.f;
  if (i < M) {
    float t = o[i]; t = (t > 0.f) ? t : 0.f;
    t += x[i];
    x[i] = t; v = t; v2 = t * t;
  }
  s1[threadIdx.x] = v; s2[threadIdx.x] = v2;
  __syncthreads();
  for (int st = 128; st > 0; st >>= 1) {
    if (threadIdx.x < st) {
      s1[threadIdx.x] += s1[threadIdx.x + st];
      s2[threadIdx.x] += s2[threadIdx.x + st];
    }
    __syncthreads();
  }
  if (threadIdx.x == 0) {
    unsafeAtomicAdd(&red[0], s1[0]);
    unsafeAtomicAdd(&red[1], s2[0]);
  }
}

// graph-mode LayerNorm apply (global mean/std, per-channel affine)
__global__ __launch_bounds__(256) void ln_apply(float* __restrict__ x,
                                                const float* __restrict__ red,
                                                const float* __restrict__ w,
                                                const float* __restrict__ b, long M) {
  long i = (long)blockIdx.x * 256 + threadIdx.x;
  if (i >= M) return;
  float mean = red[0] / (float)M;
  float var  = red[1] / (float)M - mean * mean;
  var = (var > 0.f) ? var : 0.f;
  float inv = 1.f / (sqrtf(var) + 1e-5f);
  int c = (int)(i & 31);
  x[i] = (x[i] - mean) * inv * w[c] + b[c];
}

// node encoder: out[n,c] = relu(sum_f X[n,f]*W[f,c] + b[c])
__global__ __launch_bounds__(256) void encode(const float* __restrict__ X,
                                              const float* __restrict__ W,
                                              const float* __restrict__ b,
                                              float* __restrict__ out, int N, int F) {
  int i = blockIdx.x * 256 + threadIdx.x;
  if (i >= N * CH) return;
  int n = i >> 5, c = i & 31;
  float s = b[c];
  for (int f = 0; f < F; ++f) s += X[(long)n * F + f] * W[f * CH + c];
  out[i] = (s > 0.f) ? s : 0.f;
}

// final linear 32 -> 2
__global__ __launch_bounds__(256) void lin_out(const float* __restrict__ x,
                                               const float* __restrict__ w,
                                               const float* __restrict__ b,
                                               float* __restrict__ out, int N) {
  int i = blockIdx.x * 256 + threadIdx.x;
  if (i >= N * 2) return;
  int n = i >> 1, o = i & 1;
  float s = b[o];
#pragma unroll
  for (int c = 0; c < CH; ++c) s += x[(long)n * CH + c] * w[c * 2 + o];
  out[i] = s;
}

// ---------------------------------------------------------------------------
// Host orchestration
// ---------------------------------------------------------------------------
struct Gat {
  const float *w_src, *att_src, *att_dst, *bias, *w_dst, *w_edge, *att_edge;
};
static inline const float* FP(void* const* d, int i) { return (const float*)d[i]; }
static inline Gat g7(void* const* d, int b) {
  return {FP(d,b),FP(d,b+1),FP(d,b+2),FP(d,b+3),FP(d,b+4),FP(d,b+5),FP(d,b+6)};
}
static inline Gat g4(void* const* d, int b) {
  return {FP(d,b),FP(d,b+1),FP(d,b+2),FP(d,b+3),nullptr,nullptr,nullptr};
}

static void run_gat(hipStream_t s, const Gat& p,
                    const float* xsrc, int Nsrc, const float* xdst, int Ndst,
                    const int* src, const int* dst, int E,
                    const float* ea, int Fe,
                    float* hbig, float* acc, float* a_src, float* a_dst,
                    float* denom, float* expal, float* qbuf,
                    float* out32, bool accumulate, bool shared) {
  if (!shared) {
    // dst transform only feeds a_dst -> compute then recycle hbig for hs
    gemm_32x128<<<(Ndst + 63) / 64, 256, 0, s>>>(xdst, p.w_dst ? p.w_dst : p.w_src, hbig, Ndst);
    attn_dot<<<(Ndst * 4 + 255) / 256, 256, 0, s>>>(hbig, p.att_dst, a_dst, Ndst);
    gemm_32x128<<<(Nsrc + 63) / 64, 256, 0, s>>>(xsrc, p.w_src, hbig, Nsrc);
    attn_dot<<<(Nsrc * 4 + 255) / 256, 256, 0, s>>>(hbig, p.att_src, a_src, Nsrc);
  } else {  // spent: shared weights, src==dst node set -> single GEMM
    gemm_32x128<<<(Nsrc + 63) / 64, 256, 0, s>>>(xsrc, p.w_src, hbig, Nsrc);
    attn_dot<<<(Nsrc * 4 + 255) / 256, 256, 0, s>>>(hbig, p.att_src, a_src, Nsrc);
    attn_dot<<<(Ndst * 4 + 255) / 256, 256, 0, s>>>(hbig, p.att_dst, a_dst, Ndst);
  }
  hipMemsetAsync(denom, 0, (size_t)Ndst * 4 * sizeof(float), s);
  hipMemsetAsync(acc, 0, (size_t)Ndst * HC * sizeof(float), s);
  if (ea) edge_q<<<1, 32, 0, s>>>(p.w_edge, p.att_edge, qbuf, Fe);
  edge_alpha<<<(E * 4 + 255) / 256, 256, 0, s>>>(src, dst, a_src, a_dst, ea, qbuf,
                                                 Fe, expal, denom, E);
  long tot = (long)E * 32;   // one thread per (edge, channel-quad)
  scatter_msg<<<(int)((tot + 255) / 256), 256, 0, s>>>(src, dst, hbig, expal, acc, tot);
  finalize_gat<<<(Ndst * CH + 255) / 256, 256, 0, s>>>(acc, denom, p.bias, out32,
                                                       Ndst, accumulate ? 1 : 0);
}

static void resid_ln(hipStream_t s, float* x, const float* out32, long M,
                     const float* lnw, const float* lnb, float* red) {
  hipMemsetAsync(red, 0, 2 * sizeof(float), s);
  resid_reduce<<<(int)((M + 255) / 256), 256, 0, s>>>(out32, x, red, M);
  ln_apply<<<(int)((M + 255) / 256), 256, 0, s>>>(x, red, lnw, lnb, M);
}

extern "C" void kernel_launch(void* const* d_in, const int* in_sizes, int n_in,
                              void* d_out, int out_size, void* d_ws, size_t ws_size,
                              hipStream_t stream) {
  (void)in_sizes; (void)n_in; (void)out_size; (void)ws_size;
  hipStream_t s = stream;

  // ---- inputs --------------------------------------------------------------
  const float* x_addr = FP(d_in, 0);
  const float* x_tx   = FP(d_in, 1);
  const float* ea_in  = FP(d_in, 2);
  const float* ea_out = FP(d_in, 3);
  const int* src_in  = (const int*)d_in[4];
  const int* dst_in  = (const int*)d_in[5];
  const int* src_out = (const int*)d_in[6];
  const int* dst_out = (const int*)d_in[7];
  const int* src_sp  = (const int*)d_in[8];
  const int* dst_sp  = (const int*)d_in[9];
  const float* enc_aw = FP(d_in,10); const float* enc_ab = FP(d_in,11);
  const float* enc_tw = FP(d_in,12); const float* enc_tb = FP(d_in,13);
  const float* ln_aw  = FP(d_in,14); const float* ln_ab  = FP(d_in,15);
  const float* ln_tw  = FP(d_in,16); const float* ln_tb  = FP(d_in,17);
  // conv params flattened in dict order: w_src,att_src,att_dst,bias[,w_dst][,w_edge,att_edge]
  Gat c1i = g7(d_in, 18), c1o = g7(d_in, 25), c1s = g4(d_in, 32);
  Gat c2i = g7(d_in, 36), c2o = g7(d_in, 43);
  Gat c3i = g7(d_in, 50), c3o = g7(d_in, 57), c3s = g4(d_in, 64);
  Gat c4i = g7(d_in, 68), c4o = g7(d_in, 75);
  const float* lin_w = FP(d_in, 82);
  const float* lin_b = FP(d_in, 83);

  // ---- workspace layout (floats) ------------------------------------------
  float* w = (float*)d_ws;
  float* xa     = w;                 w += (size_t)N_ADDR * CH;   // 3.2M
  float* xt     = w;                 w += (size_t)N_TX * CH;     // 4.8M
  float* hbig   = w;                 w += (size_t)N_TX * HC;     // 19.2M
  float* acc    = w;                 w += (size_t)N_TX * HC;     // 19.2M
  float* o_tx   = w;                 w += (size_t)N_TX * CH;     // 4.8M
  float* o_ad   = w;                 w += (size_t)N_ADDR * CH;   // 3.2M
  float* a_src  = w;                 w += (size_t)N_TX * 4;
  float* a_dst  = w;                 w += (size_t)N_TX * 4;
  float* denom  = w;                 w += (size_t)N_TX * 4;
  float* expal  = w;                 w += (size_t)E_IN * 4;
  float* qbuf   = w;                 w += 16;
  float* red    = w;                 w += 8;

  // ---- encoders ------------------------------------------------------------
  encode<<<(N_ADDR * CH + 255) / 256, 256, 0, s>>>(x_addr, enc_aw, enc_ab, xa, N_ADDR, F_ADDR);
  encode<<<(N_TX   * CH + 255) / 256, 256, 0, s>>>(x_tx,   enc_tw, enc_tb, xt, N_TX,   F_TX);

  // ---- 4 GAT rounds --------------------------------------------------------
  const Gat* Pi[4] = {&c1i, &c2i, &c3i, &c4i};
  const Gat* Po[4] = {&c1o, &c2o, &c3o, &c4o};
  const Gat* Ps[4] = {&c1s, nullptr, &c3s, nullptr};
  for (int r = 0; r < 4; ++r) {
    // ('addr','input','tx'): src=addr, dst=tx, edge attrs (Fe=3)
    run_gat(s, *Pi[r], xa, N_ADDR, xt, N_TX, src_in, dst_in, E_IN, ea_in, 3,
            hbig, acc, a_src, a_dst, denom, expal, qbuf, o_tx, false, false);
    // ('tx','spent','tx'): shared linear, no edge attrs; add into o_tx
    if (Ps[r])
      run_gat(s, *Ps[r], xt, N_TX, xt, N_TX, src_sp, dst_sp, E_SP, nullptr, 0,
              hbig, acc, a_src, a_dst, denom, expal, qbuf, o_tx, true, true);
    // ('tx','output','addr'): src=tx, dst=addr, edge attrs (Fe=2)
    run_gat(s, *Po[r], xt, N_TX, xa, N_ADDR, src_out, dst_out, E_OUT, ea_out, 2,
            hbig, acc, a_src, a_dst, denom, expal, qbuf, o_ad, false, false);
    // residual + graph LayerNorm (reads old x, all GATs above already consumed it)
    resid_ln(s, xa, o_ad, (long)N_ADDR * CH, ln_aw, ln_ab, red);
    resid_ln(s, xt, o_tx, (long)N_TX   * CH, ln_tw, ln_tb, red);
  }

  // ---- output head ---------------------------------------------------------
  lin_out<<<(N_ADDR * 2 + 255) / 256, 256, 0, s>>>(xa, lin_w, lin_b, (float*)d_out, N_ADDR);
}
